// SparseRouter_20289425506964
// MI455X (gfx1250) — compile-verified, matrix operand order checked by reading, not run on hardware
//
#include <hip/hip_runtime.h>
#include <hip/hip_bf16.h>
#include <math.h>

typedef __attribute__((ext_vector_type(16))) __bf16 v16bf;
typedef __attribute__((ext_vector_type(8)))  __bf16 v8bf;
typedef __attribute__((ext_vector_type(4)))  __bf16 v4bf;
typedef __attribute__((ext_vector_type(8)))  float  v8f;
typedef __attribute__((ext_vector_type(4)))  float  v4f;

#define B_TOK 8192
#define DDIM  1024
#define NEXP  32
#define CDIM  256

#define BM 128
#define BN 128
#define BK 32
#define KSTEPS (DDIM / BK)

// -------- kernel 1: per-expert inverse feature norms (32 waves, 1 block) -----
__global__ void efnorm_kernel(const float* __restrict__ ef, float* __restrict__ efInv) {
  int wave = threadIdx.x >> 5, lane = threadIdx.x & 31;
  const float* row = ef + (size_t)wave * DDIM;
  float s = 0.f;
#pragma unroll
  for (int i = 0; i < DDIM / 32; ++i) { float v = row[i * 32 + lane]; s += v * v; }
#pragma unroll
  for (int off = 16; off > 0; off >>= 1) s += __shfl_xor(s, off, 32);
  if (lane == 0) efInv[wave] = 1.f / fmaxf(sqrtf(s), 1e-8f);
}

// -------- kernel 2: score + top-4 routing, one wave per token ---------------
__global__ __launch_bounds__(256) void route_kernel(
    const float* __restrict__ x, const float* __restrict__ ef,
    const float* __restrict__ trust, const float* __restrict__ stal,
    const float* __restrict__ efInv, int* __restrict__ counts,
    int* __restrict__ lists) {
  int wave = threadIdx.x >> 5, lane = threadIdx.x & 31;
  int b = blockIdx.x * 8 + wave;

  const float* xr = x + (size_t)b * DDIM;
  float xf[32];
  float ss = 0.f;
#pragma unroll
  for (int i = 0; i < 32; ++i) { float v = xr[i * 32 + lane]; xf[i] = v; ss += v * v; }
#pragma unroll
  for (int off = 16; off > 0; off >>= 1) ss += __shfl_xor(ss, off, 32);
  float xinv = 1.f / fmaxf(sqrtf(ss), 1e-8f);

  float mydot = 0.f;
  for (int e = 0; e < NEXP; ++e) {
    const float* er = ef + (size_t)e * DDIM;
    float d = 0.f;
#pragma unroll
    for (int i = 0; i < 32; ++i) d = fmaf(xf[i], er[i * 32 + lane], d);
#pragma unroll
    for (int off = 16; off > 0; off >>= 1) d += __shfl_xor(d, off, 32);
    if (lane == e) mydot = d;
  }

  float cosv  = mydot * xinv * efInv[lane];
  float score = 0.4f * trust[lane] + 0.4f * (cosv + 1.f) * 0.5f +
                0.2f * fmaxf(0.f, 1.f - stal[lane]);

  // 4-round argmax (ties -> lowest index, matching top_k)
  int   myidx = 0;
  float s = score;
#pragma unroll
  for (int t = 0; t < 4; ++t) {
    float bs = s; int bi = lane;
#pragma unroll
    for (int off = 16; off > 0; off >>= 1) {
      float os = __shfl_xor(bs, off, 32);
      int   oi = __shfl_xor(bi, off, 32);
      if (os > bs || (os == bs && oi < bi)) { bs = os; bi = oi; }
    }
    if (lane == t)  myidx = bi;
    if (lane == bi) s = -3.4e38f;
  }

  if (lane < 4) {
    int pos = atomicAdd(&counts[myidx], 1);
    lists[myidx * B_TOK + pos] = b;
  }
}

// -------- kernel 2.5: f32 -> bf16 row-major pack (x and W) ------------------
__global__ __launch_bounds__(256) void pack_bf16_kernel(
    const float* __restrict__ src, __bf16* __restrict__ dst, int n8) {
  int i = blockIdx.x * 256 + threadIdx.x;
  if (i >= n8) return;
  const v4f* s = (const v4f*)(src + (size_t)i * 8);
  v4bf* d = (v4bf*)(dst + (size_t)i * 8);
  d[0] = __builtin_convertvector(s[0], v4bf);
  d[1] = __builtin_convertvector(s[1], v4bf);
}

// -------- kernel 3: grouped gather-GEMM, bf16 WMMA, double-buffered LDS -----
// WMMA 16-bit fragment layout (ISA 7.12.2):
//   lane     = (mn&15) | (((k>>3)&1)<<4)
//   halfword = (k&7)   | ((k>>4)<<3)
// An 8-wide aligned k-group maps to one contiguous 16B chunk per lane.
template <bool PACKED>
__global__ __launch_bounds__(256) void moe_gemm_kernel(
    const void* __restrict__ xsrc, const void* __restrict__ wsrc,
    const float* __restrict__ bias, const int* __restrict__ counts,
    const int* __restrict__ lists, float* __restrict__ out) {
  __shared__ __align__(32) __bf16 sA[2][BM * BK];
  __shared__ __align__(32) __bf16 sB[2][BN * BK];
  __shared__ int sTok[BM];

  const int e   = blockIdx.z;
  const int cnt = counts[e];
  const int m0  = blockIdx.y * BM;
  if (m0 >= cnt) return;                    // uniform: EXEC stays full for WMMA
  const int c0   = blockIdx.x * BN;
  const int tid  = threadIdx.x;
  const int lane = tid & 31, waveId = tid >> 5;
  const int wm0  = (waveId & 3) * 32;       // 4 waves along M
  const int wn0  = (waveId >> 2) * 64;      // 2 waves along N

  if (tid < BM) {
    int g = m0 + tid;
    sTok[tid] = (g < cnt) ? lists[e * B_TOK + g] : lists[e * B_TOK + m0];
  }
  __syncthreads();

  auto stage = [&](int ks, int buf) {
    const int k0 = ks * BK;
    if constexpr (PACKED) {
      const __bf16* xb = (const __bf16*)xsrc;
      const __bf16* Wb = (const __bf16*)wsrc + ((size_t)e * CDIM + c0) * DDIM;
#pragma unroll
      for (int t = 0; t < 2; ++t) {
        int lin = tid + t * 256;                        // 0..511
        int m   = lin >> 2;                             // 0..127
        int kg  = (lin & 3) << 3;                       // 0,8,16,24
        int ofs = (((m >> 4) * 32) + ((m & 15) | ((kg & 8) << 1))) * 16 +
                  ((kg & 16) >> 1);
        *(v8bf*)&sA[buf][ofs] =
            *(const v8bf*)(xb + (size_t)sTok[m] * DDIM + k0 + kg);
        const __bf16* wp = Wb + (size_t)m * DDIM + k0 + kg;
        *(v8bf*)&sB[buf][ofs] = *(const v8bf*)wp;
        __builtin_prefetch(wp + BK, 0, 1);              // next K-slab of W
      }
    } else {
      const float* xf = (const float*)xsrc;
      const float* Wf = (const float*)wsrc + ((size_t)e * CDIM + c0) * DDIM;
#pragma unroll
      for (int t = 0; t < 4; ++t) {
        int lin = tid + t * 256;                        // 0..1023
        int m   = lin >> 3;                             // 0..127
        int kg  = (lin & 7) << 2;                       // 0,4,...,28
        int ofs = (((m >> 4) * 32) + ((m & 15) | ((kg & 8) << 1))) * 16 +
                  ((kg & 7) | ((kg >> 4) << 3));
        v4f va = *(const v4f*)(xf + (size_t)sTok[m] * DDIM + k0 + kg);
        *(v4bf*)&sA[buf][ofs] = __builtin_convertvector(va, v4bf);
        const float* wp = Wf + (size_t)m * DDIM + k0 + kg;
        *(v4bf*)&sB[buf][ofs] = __builtin_convertvector(*(const v4f*)wp, v4bf);
        __builtin_prefetch(wp + BK, 0, 1);
      }
    }
  };

  v8f acc[2][4];
#pragma unroll
  for (int i = 0; i < 2; ++i)
#pragma unroll
    for (int j = 0; j < 4; ++j)
#pragma unroll
      for (int r = 0; r < 8; ++r) acc[i][j][r] = 0.f;

  int buf = 0;
  stage(0, 0);
  __syncthreads();

  for (int ks = 0; ks < KSTEPS; ++ks) {
    if (ks + 1 < KSTEPS) stage(ks + 1, buf ^ 1);        // overlap with WMMA below

    v16bf af[2], bfr[4];
#pragma unroll
    for (int i = 0; i < 2; ++i)
      af[i] = *(const v16bf*)&sA[buf][(((wm0 >> 4) + i) * 32 + lane) * 16];
#pragma unroll
    for (int j = 0; j < 4; ++j)
      bfr[j] = *(const v16bf*)&sB[buf][(((wn0 >> 4) + j) * 32 + lane) * 16];

#pragma unroll
    for (int i = 0; i < 2; ++i)
#pragma unroll
      for (int j = 0; j < 4; ++j)
        acc[i][j] = __builtin_amdgcn_wmma_f32_16x16x32_bf16(
            false, af[i], false, bfr[j], (short)0, acc[i][j], false, false);

    __syncthreads();
    buf ^= 1;
  }

  // epilogue: D layout (VGPR r -> M = r + 8*(lane>=16); N = lane&15)
  const int nloc = lane & 15;
  const int half = lane >> 4;
#pragma unroll
  for (int j = 0; j < 4; ++j) {
    int col = c0 + wn0 + j * 16 + nloc;
    float bv = bias[e * CDIM + col];
#pragma unroll
    for (int i = 0; i < 2; ++i) {
#pragma unroll
      for (int r = 0; r < 8; ++r) {
        int row = wm0 + i * 16 + r + 8 * half;
        if (m0 + row < cnt) {
          atomicAdd(out + (size_t)sTok[row] * CDIM + col,
                    0.25f * (acc[i][j][r] + bv));
        }
      }
    }
  }
}

extern "C" void kernel_launch(void* const* d_in, const int* in_sizes, int n_in,
                              void* d_out, int out_size, void* d_ws, size_t ws_size,
                              hipStream_t stream) {
  (void)in_sizes; (void)n_in;
  const float* x     = (const float*)d_in[0];
  const float* ef    = (const float*)d_in[1];
  const float* trust = (const float*)d_in[2];
  const float* stal  = (const float*)d_in[3];
  const float* W     = (const float*)d_in[4];
  const float* bias  = (const float*)d_in[5];
  // d_in[6] = k, fixed at 4 (weight 0.25 baked in)
  float* out = (float*)d_out;

  char*  ws     = (char*)d_ws;
  int*   counts = (int*)ws;                        // 32 ints
  float* efInv  = (float*)(ws + 256);              // 32 floats
  int*   lists  = (int*)(ws + 512);                // 32*8192 ints (1 MB)

  const size_t lists_end = 512 + (size_t)NEXP * B_TOK * sizeof(int);
  const size_t off_xbf   = (lists_end + 255) & ~(size_t)255;
  const size_t xbf_bytes = (size_t)B_TOK * DDIM * 2;            // 16 MB
  const size_t off_wbf   = off_xbf + xbf_bytes;
  const size_t wbf_bytes = (size_t)NEXP * CDIM * DDIM * 2;      // 16 MB
  const bool packed = (ws_size >= off_wbf + wbf_bytes);

  hipMemsetAsync(counts, 0, NEXP * sizeof(int), stream);
  hipMemsetAsync(out, 0, (size_t)out_size * sizeof(float), stream);

  efnorm_kernel<<<1, 1024, 0, stream>>>(ef, efInv);
  route_kernel<<<B_TOK / 8, 256, 0, stream>>>(x, ef, trust, stal, efInv, counts, lists);

  const dim3 ggrid(CDIM / BN, B_TOK / BM, NEXP);
  if (packed) {
    __bf16* xbf = (__bf16*)(ws + off_xbf);
    __bf16* wbf = (__bf16*)(ws + off_wbf);
    const int n8x = B_TOK * DDIM / 8;                 // 1,048,576
    const int n8w = NEXP * CDIM * DDIM / 8;           // 1,048,576
    pack_bf16_kernel<<<(n8x + 255) / 256, 256, 0, stream>>>(x, xbf, n8x);
    pack_bf16_kernel<<<(n8w + 255) / 256, 256, 0, stream>>>(W, wbf, n8w);
    moe_gemm_kernel<true><<<ggrid, 256, 0, stream>>>(xbf, wbf, bias, counts, lists, out);
  } else {
    moe_gemm_kernel<false><<<ggrid, 256, 0, stream>>>(x, W, bias, counts, lists, out);
  }
}
